// VQ_10831907520890
// MI455X (gfx1250) — matrix operand order repaired and checked by reading, not compiled, not run on hardware
//
#include <hip/hip_runtime.h>
#include <hip/hip_bf16.h>

typedef __attribute__((ext_vector_type(16))) __bf16 v16bf;
typedef __attribute__((ext_vector_type(8)))  float  v8f;

#define N_ROWS 32768
#define DIM    256
#define K_CODE 4096
#define KCHUNKS (K_CODE / 16)   // 256 chunks of 16 columns

union U16x16 {
    v16bf v;
    unsigned short s[16];
    uint4 q[2];
};

// round-to-nearest-even f32 -> bf16 (bit pattern)
static __device__ __forceinline__ unsigned short f2bf(float f) {
    unsigned int u = __float_as_uint(f);
    u += 0x7FFFu + ((u >> 16) & 1u);
    return (unsigned short)(u >> 16);
}

static __device__ __forceinline__ void wait_asynccnt0() {
#if __has_builtin(__builtin_amdgcn_s_wait_asynccnt)
    __builtin_amdgcn_s_wait_asynccnt(0);
#else
    asm volatile("s_wait_asynccnt 0x0" ::: "memory");
#endif
}

// Async DMA: copy 32B per lane (two 16B loads 4096B apart) global->LDS (ASYNCcnt-tracked, no VGPR data).
// INST_OFFSET is added to BOTH the LDS and global addresses (ISA 08 §4.4).
static __device__ __forceinline__ void async_copy_b128(unsigned lds_off,
                                                       const void* gaddr) {
    unsigned long long g = (unsigned long long)(uintptr_t)gaddr;
    asm volatile("global_load_async_to_lds_b128 %0, %1, off"
                 :: "v"(lds_off), "v"(g) : "memory");
    asm volatile("global_load_async_to_lds_b128 %0, %1, off offset:4096"
                 :: "v"(lds_off), "v"(g) : "memory");
}

// ---------------------------------------------------------------------------
// Kernel 1: negHalf[k] = -0.5 * sum_d E[d,k]^2
// ---------------------------------------------------------------------------
__global__ void vq_norm_kernel(const float* __restrict__ E, float* __restrict__ negHalf) {
    int k = blockIdx.x * blockDim.x + threadIdx.x;   // 4096 threads
    float s = 0.0f;
    #pragma unroll 8
    for (int d = 0; d < DIM; ++d) {
        float v = E[d * K_CODE + k];
        s += v * v;
    }
    negHalf[k] = -0.5f * s;
}

// ---------------------------------------------------------------------------
// Kernel 2: pack E (f32 [D,K]) into bf16 WMMA B-fragments.
// Chunk c covers columns [16c,16c+16). Fragment i covers K-dim (D) slice
// [32i,32i+32). Per-lane 16 halves stored contiguously (32B):
//   lane l: column n = l&15, kk = (l<16 ? 0 : 8) + (t&7) + (t>>3)*16  (t = 0..15)
//   addr(half) = ((c*8 + i)*32 + l)*16 + t
// ---------------------------------------------------------------------------
__global__ void vq_pack_kernel(const float* __restrict__ E, unsigned short* __restrict__ Bpack) {
    __shared__ float lds[DIM * 16];   // 16KB: [d][n]
    int c = blockIdx.x;               // 256 chunks
    int t0 = threadIdx.x;             // 256 threads

    #pragma unroll
    for (int r = 0; r < 16; ++r) {
        int idx = r * 256 + t0;       // 0..4095
        int d = idx >> 4;
        int n = idx & 15;
        lds[idx] = E[d * K_CODE + 16 * c + n];   // 64B-coalesced groups
    }
    __syncthreads();

    #pragma unroll
    for (int w = 0; w < 16; ++w) {
        int idx = w * 256 + t0;       // half index within chunk, 0..4095
        int i   = idx >> 9;           // fragment 0..7
        int rem = idx & 511;
        int l   = rem >> 4;           // lane 0..31
        int tt  = rem & 15;           // element 0..15
        int n   = l & 15;
        int d   = 32 * i + ((l < 16) ? 0 : 8) + (tt & 7) + (tt >> 3) * 16;
        Bpack[c * 4096 + idx] = f2bf(lds[d * 16 + n]);  // fully coalesced store
    }
}

// ---------------------------------------------------------------------------
// Kernel 3: main fused GEMM + argmax + gather + loss partial sums.
// 256 blocks x 256 threads (8 waves). Wave w owns rows [blk*128+16w, +16).
// B chunks are prefetched with async global->LDS DMA, double buffered.
// ---------------------------------------------------------------------------
__global__ void __launch_bounds__(256)
vq_main_kernel(const float* __restrict__ x,
               const float* __restrict__ E,
               const unsigned short* __restrict__ Bpack,
               const float* __restrict__ negHalf,
               float* __restrict__ out,
               float* __restrict__ sumAcc) {
    __shared__ uint4 ldsB[2][512];    // double-buffered 8KB B chunks

    const int tid  = threadIdx.x;
    const int wave = tid >> 5;
    const int lane = tid & 31;
    const int hl   = lane >> 4;       // half-wave: 0 -> rows 0..7, 1 -> rows 8..15
    const int col  = lane & 15;       // column within 16-wide chunk

    const int rowBase = blockIdx.x * 128 + wave * 16;

    // per-thread addresses for the async staging (16B granularity)
    const unsigned ldsOff0 = (unsigned)(uintptr_t)(&ldsB[0][tid]);
    const unsigned ldsOff1 = (unsigned)(uintptr_t)(&ldsB[1][tid]);

    // ---- kick off prefetch of chunk 0 into buffer 0 ----
    async_copy_b128(ldsOff0, (const uint4*)Bpack + tid);

    // ---- Load A tile: 16 rows x 256 D of x as bf16, WMMA A layout ----
    // lane l holds row M = l&15; K runs: [32i + 8*hl, +8) and [32i+16+8*hl, +8)
    const int m = rowBase + col;
    const float* px = x + (size_t)m * DIM;
    v16bf a[8];
    #pragma unroll
    for (int i = 0; i < 8; ++i) {
        int d0 = 32 * i + hl * 8;
        float4 f0 = *(const float4*)(px + d0);
        float4 f1 = *(const float4*)(px + d0 + 4);
        float4 f2 = *(const float4*)(px + d0 + 16);
        float4 f3 = *(const float4*)(px + d0 + 20);
        U16x16 au;
        au.s[0]  = f2bf(f0.x); au.s[1]  = f2bf(f0.y); au.s[2]  = f2bf(f0.z); au.s[3]  = f2bf(f0.w);
        au.s[4]  = f2bf(f1.x); au.s[5]  = f2bf(f1.y); au.s[6]  = f2bf(f1.z); au.s[7]  = f2bf(f1.w);
        au.s[8]  = f2bf(f2.x); au.s[9]  = f2bf(f2.y); au.s[10] = f2bf(f2.z); au.s[11] = f2bf(f2.w);
        au.s[12] = f2bf(f3.x); au.s[13] = f2bf(f3.y); au.s[14] = f2bf(f3.z); au.s[15] = f2bf(f3.w);
        a[i] = au.v;
    }

    float best[8];
    int   bidx[8];
    #pragma unroll
    for (int j = 0; j < 8; ++j) { best[j] = -3.4e38f; bidx[j] = 0; }

    // ---- Sweep all K chunks (software-pipelined async staging) ----
    for (int c = 0; c < KCHUNKS; ++c) {
        const int buf = c & 1;

        // my async writes for chunk c have landed; barrier => everyone's have,
        // and everyone finished reading buf^1 in iteration c-1.
        wait_asynccnt0();
        __syncthreads();

        if (c + 1 < KCHUNKS) {
            async_copy_b128(buf ? ldsOff0 : ldsOff1,
                            (const uint4*)Bpack + (size_t)(c + 1) * 512 + tid);
        }

        float nh = negHalf[c * 16 + col];

        // batch-load all 8 B fragments, then chain the 8 WMMAs back-to-back
        const U16x16* bl = (const U16x16*)&ldsB[buf][0];
        U16x16 bu[8];
        #pragma unroll
        for (int i = 0; i < 8; ++i)
            bu[i] = bl[i * 32 + lane];   // 32B ds read, lane-exact fragment

        v8f acc = {};
        #pragma unroll
        for (int i = 0; i < 8; ++i)
            acc = __builtin_amdgcn_wmma_f32_16x16x32_bf16(
                false, a[i], false, bu[i].v, (short)0, acc, false, false);

        // score = x.e - 0.5*||e||^2 ; strict '>' keeps lowest index on ties
        #pragma unroll
        for (int j = 0; j < 8; ++j) {
            float t = acc[j] + nh;
            if (t > best[j]) { best[j] = t; bidx[j] = c * 16 + col; }
        }
    }

    // ---- cross-lane argmax within each half-wave (masks stay < 16) ----
    #pragma unroll
    for (int msk = 1; msk < 16; msk <<= 1) {
        #pragma unroll
        for (int j = 0; j < 8; ++j) {
            float ob = __shfl_xor(best[j], msk, 32);
            int   oi = __shfl_xor(bidx[j], msk, 32);
            if (ob > best[j] || (ob == best[j] && oi < bidx[j])) {
                best[j] = ob; bidx[j] = oi;
            }
        }
    }

    // ---- gather winning codebook columns (exact f32), write quantized,
    //      accumulate squared error ----
    float sq = 0.0f;
    #pragma unroll
    for (int j = 0; j < 8; ++j) {
        int k = bidx[j];                       // uniform across half-wave
        int r = rowBase + hl * 8 + j;          // global row
        #pragma unroll
        for (int d = col; d < DIM; d += 16) {  // 16 lanes cover 256 cols
            float e  = E[(size_t)d * K_CODE + k];
            float xv = x[(size_t)r * DIM + d];
            out[(size_t)r * DIM + d] = e;      // quantized == e_k numerically
            float df = xv - e;
            sq += df * df;
        }
    }

    // wave-wide sum, one atomic per wave
    #pragma unroll
    for (int msk = 1; msk < 32; msk <<= 1)
        sq += __shfl_xor(sq, msk, 32);
    if (lane == 0)
        atomicAdd(sumAcc, sq);
}

// ---------------------------------------------------------------------------
// Kernel 4: finalize the two scalar losses
// ---------------------------------------------------------------------------
__global__ void vq_finalize_kernel(const float* __restrict__ sumAcc,
                                   const int* __restrict__ beta,
                                   float* __restrict__ out) {
    float mean = *sumAcc * (1.0f / ((float)N_ROWS * (float)DIM));
    out[(size_t)N_ROWS * DIM]     = mean * (float)(*beta);  // commitment_loss * beta
    out[(size_t)N_ROWS * DIM + 1] = mean;                   // vq_loss
}

extern "C" void kernel_launch(void* const* d_in, const int* in_sizes, int n_in,
                              void* d_out, int out_size, void* d_ws, size_t ws_size,
                              hipStream_t stream) {
    (void)in_sizes; (void)n_in; (void)out_size; (void)ws_size;
    const float* x    = (const float*)d_in[0];
    const float* E    = (const float*)d_in[1];
    const int*   beta = (const int*)d_in[2];
    float* out = (float*)d_out;

    // ws layout: [0,2MB) packed bf16 B-fragments; [2MB,+16KB) negHalf norms; then 4B sum
    unsigned short* Bpack = (unsigned short*)d_ws;
    float* negHalf = (float*)((char*)d_ws + (size_t)2 * 1024 * 1024);
    float* sumAcc  = (float*)((char*)d_ws + (size_t)2 * 1024 * 1024 + 16384);

    hipMemsetAsync(sumAcc, 0, sizeof(float), stream);
    vq_norm_kernel<<<K_CODE / 256, 256, 0, stream>>>(E, negHalf);
    vq_pack_kernel<<<KCHUNKS, 256, 0, stream>>>(E, Bpack);
    vq_main_kernel<<<N_ROWS / 128, 256, 0, stream>>>(x, E, Bpack, negHalf, out, sumAcc);
    vq_finalize_kernel<<<1, 1, 0, stream>>>(sumAcc, beta, out);
}
